// Pointnet2Partial_15324443312439
// MI455X (gfx1250) — matrix operand order repaired and checked by reading, not compiled, not run on hardware
//
#include <hip/hip_runtime.h>
#include <hip/hip_bf16.h>
#include <stdint.h>

typedef __attribute__((ext_vector_type(16))) _Float16 v16h;
typedef __attribute__((ext_vector_type(8)))  float    v8f;

#define KNB 64
#define BN_EPS 1e-5f

// ---------------------------------------------------------------------------
// utility kernels
// ---------------------------------------------------------------------------
__global__ void zero_f32(float* p, int n) {
  int i = blockIdx.x * blockDim.x + threadIdx.x;
  if (i < n) p[i] = 0.0f;
}

__global__ void copy_f32(const float* __restrict__ s, float* __restrict__ d, int n) {
  int i = blockIdx.x * blockDim.x + threadIdx.x;
  if (i < n) d[i] = s[i];
}

// ---------------------------------------------------------------------------
// Farthest point sampling: one block per batch, all points staged in LDS,
// min-distance state in registers, block-wide argmax per step.
// blockDim.x == 1024. Writes centers directly (gather fused).
// ---------------------------------------------------------------------------
__global__ void fps_kernel(const float* __restrict__ pos, int N, int M,
                           float* __restrict__ centers) {
  __shared__ float px[4096];
  __shared__ float py[4096];
  __shared__ float pz[4096];
  __shared__ float rv[1024];
  __shared__ int   ri[1024];
  __shared__ float cur[3];

  const int b   = blockIdx.x;
  const int tid = threadIdx.x;
  const float* p = pos + (size_t)b * N * 3;

  for (int i = tid; i < N; i += 1024) {
    px[i] = p[3 * i + 0];
    py[i] = p[3 * i + 1];
    pz[i] = p[3 * i + 2];
  }

  float mind[4];
  const int nchunk = (N + 1023) >> 10;
  for (int c = 0; c < 4; ++c) mind[c] = 1e10f;

  __syncthreads();
  if (tid == 0) {
    cur[0] = px[0]; cur[1] = py[0]; cur[2] = pz[0];
    centers[((size_t)b * M) * 3 + 0] = px[0];
    centers[((size_t)b * M) * 3 + 1] = py[0];
    centers[((size_t)b * M) * 3 + 2] = pz[0];
  }
  __syncthreads();

  for (int m = 1; m < M; ++m) {
    const float cx = cur[0], cy = cur[1], cz = cur[2];
    float best = -1.0f;
    int   bi   = 0;
    for (int c = 0; c < nchunk; ++c) {
      int i = tid + (c << 10);
      if (i < N) {
        float dx = px[i] - cx, dy = py[i] - cy, dz = pz[i] - cz;
        float d  = dx * dx + dy * dy + dz * dz;
        float mn = fminf(mind[c], d);
        mind[c] = mn;
        if (mn > best) { best = mn; bi = i; }
      }
    }
    rv[tid] = best; ri[tid] = bi;
    __syncthreads();
    for (int s = 512; s > 0; s >>= 1) {
      if (tid < s) {
        if (rv[tid + s] > rv[tid]) { rv[tid] = rv[tid + s]; ri[tid] = ri[tid + s]; }
      }
      __syncthreads();
    }
    if (tid == 0) {
      int sel = ri[0];
      cur[0] = px[sel]; cur[1] = py[sel]; cur[2] = pz[sel];
      centers[((size_t)b * M + m) * 3 + 0] = px[sel];
      centers[((size_t)b * M + m) * 3 + 1] = py[sel];
      centers[((size_t)b * M + m) * 3 + 2] = pz[sel];
    }
    __syncthreads();
  }
}

// ---------------------------------------------------------------------------
// Radius grouping: one 64-thread block per (batch, center). Claims up to KNB
// neighbor slots via LDS atomic counter; accumulates global valid-edge count.
// ---------------------------------------------------------------------------
__global__ void group_kernel(const float* __restrict__ pos,
                             const float* __restrict__ centers,
                             int N, int M, float r2,
                             int* __restrict__ nidx, int* __restrict__ mask,
                             float* __restrict__ cntg) {
  __shared__ int cnt;
  const int bm = blockIdx.x;          // b*M + m
  const int b  = bm / M;
  if (threadIdx.x == 0) cnt = 0;
  __syncthreads();

  const float* p = pos + (size_t)b * N * 3;
  const float* c = centers + (size_t)bm * 3;
  const float cx = c[0], cy = c[1], cz = c[2];
  int* nrow = nidx + (size_t)bm * KNB;

  for (int i = threadIdx.x; i < N; i += blockDim.x) {
    float dx = p[3 * i + 0] - cx;
    float dy = p[3 * i + 1] - cy;
    float dz = p[3 * i + 2] - cz;
    float d  = dx * dx + dy * dy + dz * dz;
    if (d <= r2) {
      int slot = atomicAdd(&cnt, 1);
      if (slot < KNB) nrow[slot] = i;
    }
  }
  __syncthreads();
  int total = cnt < KNB ? cnt : KNB;
  int j = threadIdx.x;
  if (j < KNB) {
    mask[(size_t)bm * KNB + j] = (j < total) ? 1 : 0;
    if (j >= total) nrow[j] = 0;
  }
  if (threadIdx.x == 0) atomicAdd(cntg, (float)total);
}

// ---------------------------------------------------------------------------
// Fragment-swizzle helper (CDNA5 WMMA 16x16x32 f16 A layout, ISA 7.12.2).
// A (16x32, MxK):  lane L elem e -> row = L%16, k = (L>=16?8:0) + (e<8? e : 8+e)
//   inverse: L = r + ((k&8)?16:0), e = (k>=16?8:0) + (k&7)
// ---------------------------------------------------------------------------
__device__ __forceinline__ size_t a_swz_index(long long row, int col, int nKc) {
  long long t = row >> 4;
  int r  = (int)(row & 15);
  int cc = col >> 5;
  int kl = col & 31;
  int L  = r + ((kl & 8) ? 16 : 0);
  int e  = ((kl >= 16) ? 8 : 0) + (kl & 7);
  return ((size_t)(t * nKc + cc) * 32 + L) * 16 + e;
}

// ---------------------------------------------------------------------------
// Weight convert: f32 [Cin,Cout] -> f16 B-fragment-swizzled, zero-padded K.
// B (32x16, KxN): lane L elem e -> n = L%16, k = (L>=16?8:0) + (e<8? e : 8+e)
// ---------------------------------------------------------------------------
__global__ void convert_w(const float* __restrict__ W, int Cin, int Cout, int Cpad,
                          _Float16* __restrict__ Wswz) {
  int idx = blockIdx.x * blockDim.x + threadIdx.x;
  int tot = Cout * Cpad;
  if (idx >= tot) return;
  int n = idx / Cpad;
  int k = idx % Cpad;
  float v = (k < Cin) ? W[(size_t)k * Cout + n] : 0.0f;
  int nKc = Cpad >> 5;
  int j  = n >> 4;
  int c  = k >> 5;
  int kl = k & 31;
  int L  = (n & 15) + ((kl & 8) ? 16 : 0);
  int e  = ((kl >= 16) ? 8 : 0) + (kl & 7);
  Wswz[((size_t)(j * nKc + c) * 32 + L) * 16 + e] = (_Float16)v;
}

// ---------------------------------------------------------------------------
// Build edge-feature matrix A (f16, A-fragment-swizzled, K padded to Cpad):
// cols [0,featC): gathered features; [featC,featC+3): pos_j - center; rest 0.
// Invalid edges are fully zeroed.
// ---------------------------------------------------------------------------
__global__ void build_edges(const float* __restrict__ feat, int featC,
                            const float* __restrict__ pos,
                            const float* __restrict__ centers,
                            const int* __restrict__ nidx,
                            const int* __restrict__ mask,
                            int Mloc, int Npts, int Cpad, long long Etot,
                            _Float16* __restrict__ Aswz) {
  long long total = Etot * (long long)Cpad;
  int nKc = Cpad >> 5;
  for (long long t = blockIdx.x * (long long)blockDim.x + threadIdx.x;
       t < total; t += (long long)gridDim.x * blockDim.x) {
    long long e = t / Cpad;
    int col = (int)(t % Cpad);
    float val = 0.0f;
    if (mask[e]) {
      long long bm = e / KNB;
      int b = (int)(bm / Mloc);
      int pj = nidx[e];
      if (col < featC) {
        val = feat[((size_t)b * Npts + pj) * featC + col];
      } else if (col < featC + 3) {
        int d = col - featC;
        val = pos[((size_t)b * Npts + pj) * 3 + d] - centers[(size_t)bm * 3 + d];
      }
    }
    Aswz[a_swz_index(e, col, nKc)] = (_Float16)val;
  }
}

// ---------------------------------------------------------------------------
// WMMA GEMM: H[E,Cout] = relu(A[E,Cpad] * W[Cpad,Cout] + bias).
// One wave per 16-row tile. The wave's A fragments for the whole K dimension
// are held in registers (NKC*8 VGPRs) and reused across all NN column tiles,
// so A streams from HBM exactly once; B (weights) is tiny and cache-resident.
// Fully unrolled via template params -> dense back-to-back v_wmma issue.
// ---------------------------------------------------------------------------
template <int NKC, int NN>
__global__ void __launch_bounds__(32)
gemm_wmma_t(const v16h* __restrict__ Aswz, const v16h* __restrict__ Bswz,
            const float* __restrict__ bias, _Float16* __restrict__ H) {
  const int t = blockIdx.x;               // row tile
  const int L = threadIdx.x;              // lane 0..31
  const int Cout = NN * 16;

  v16h a[NKC];
  const v16h* Ab = Aswz + (size_t)t * NKC * 32 + L;
#pragma unroll
  for (int kc = 0; kc < NKC; ++kc) a[kc] = Ab[kc * 32];

  const int rbase = (t << 4) + ((L >= 16) ? 8 : 0);
  const int nlo   = L & 15;

#pragma unroll
  for (int j = 0; j < NN; ++j) {
    const v16h* Bb = Bswz + (size_t)j * NKC * 32 + L;
    v8f c = {};
#pragma unroll
    for (int kc = 0; kc < NKC; ++kc) {
      v16h b = Bb[kc * 32];
      c = __builtin_amdgcn_wmma_f32_16x16x32_f16(
          /*neg_a=*/false, a[kc], /*neg_b=*/false, b,
          /*c_mod=*/(short)0, c, /*reuse_a=*/false, /*reuse_b=*/false);
    }
    const int n  = (j << 4) + nlo;
    const float bv = bias[n];
#pragma unroll
    for (int v = 0; v < 8; ++v) {
      float val = c[v] + bv;
      val = val > 0.0f ? val : 0.0f;               // ReLU
      H[(size_t)(rbase + v) * Cout + n] = (_Float16)val;
    }
  }
}

// ---------------------------------------------------------------------------
// Masked BN statistics: per-channel sum/sumsq of valid edges.
// LDS partial accumulation (ds_add_f32) then global atomics.
// ---------------------------------------------------------------------------
__global__ void bn_stats(const _Float16* __restrict__ H, const int* __restrict__ mrow,
                         long long E, int C,
                         float* __restrict__ gsum, float* __restrict__ gsq) {
  __shared__ float ssum[256];
  __shared__ float ssq[256];
  for (int i = threadIdx.x; i < C; i += blockDim.x) { ssum[i] = 0.0f; ssq[i] = 0.0f; }
  __syncthreads();

  long long total = E * (long long)C;
  for (long long t = blockIdx.x * (long long)blockDim.x + threadIdx.x;
       t < total; t += (long long)gridDim.x * blockDim.x) {
    long long row = t / C;
    int col = (int)(t % C);
    if (mrow[row]) {
      float v = (float)H[t];
      atomicAdd(&ssum[col], v);
      atomicAdd(&ssq[col], v * v);
    }
  }
  __syncthreads();
  for (int i = threadIdx.x; i < C; i += blockDim.x) {
    atomicAdd(&gsum[i], ssum[i]);
    atomicAdd(&gsq[i], ssq[i]);
  }
}

// ---------------------------------------------------------------------------
// BN apply (intermediate layers): normalize, zero invalid rows, and emit the
// next layer's A matrix directly in fragment-swizzled f16 form.
// ---------------------------------------------------------------------------
__global__ void bn_apply(const _Float16* __restrict__ H, const int* __restrict__ mrow,
                         const float* __restrict__ gsum, const float* __restrict__ gsq,
                         const float* __restrict__ cntp,
                         const float* __restrict__ gamma, const float* __restrict__ beta,
                         long long E, int C, _Float16* __restrict__ Anext) {
  const float cnt = fmaxf(cntp[0], 1.0f);
  const int nKc = C >> 5;
  long long total = E * (long long)C;
  for (long long t = blockIdx.x * (long long)blockDim.x + threadIdx.x;
       t < total; t += (long long)gridDim.x * blockDim.x) {
    long long row = t / C;
    int col = (int)(t % C);
    float mean = gsum[col] / cnt;
    float var  = fmaxf(gsq[col] / cnt - mean * mean, 0.0f);
    float out  = ((float)H[t] - mean) * rsqrtf(var + BN_EPS) * gamma[col] + beta[col];
    if (!mrow[row]) out = 0.0f;
    Anext[a_swz_index(row, col, nKc)] = (_Float16)out;
  }
}

// ---------------------------------------------------------------------------
// BN apply + masked max aggregation (last layer of a module).
// One block per (b,m); threads stride over channels, loop over K neighbors.
// ---------------------------------------------------------------------------
__global__ void bn_apply_max(const _Float16* __restrict__ H, const int* __restrict__ mask,
                             const float* __restrict__ gsum, const float* __restrict__ gsq,
                             const float* __restrict__ cntp,
                             const float* __restrict__ gamma, const float* __restrict__ beta,
                             int C, float* __restrict__ out) {
  const int bm = blockIdx.x;
  const float cnt = fmaxf(cntp[0], 1.0f);
  for (int col = threadIdx.x; col < C; col += blockDim.x) {
    float mean = gsum[col] / cnt;
    float var  = fmaxf(gsq[col] / cnt - mean * mean, 0.0f);
    float sc   = rsqrtf(var + BN_EPS) * gamma[col];
    float sh   = beta[col];
    float mx = -INFINITY;
    for (int k = 0; k < KNB; ++k) {
      long long e = (long long)bm * KNB + k;
      if (mask[e]) {
        float v = ((float)H[e * C + col] - mean) * sc + sh;
        mx = fmaxf(mx, v);
      }
    }
    out[(size_t)bm * C + col] = mx;
  }
}

// ---------------------------------------------------------------------------
// GEMM dispatch over the 6 static layer shapes
// ---------------------------------------------------------------------------
static void launch_gemm(int layer, long long E,
                        const v16h* A, const v16h* B, const float* bias,
                        _Float16* H, hipStream_t stream) {
  int blocks = (int)(E >> 4);
  switch (layer) {
    case 0: gemm_wmma_t<3, 4><<<blocks, 32, 0, stream>>>(A, B, bias, H); break;   //  96 ->  64
    case 1: gemm_wmma_t<2, 4><<<blocks, 32, 0, stream>>>(A, B, bias, H); break;   //  64 ->  64
    case 2: gemm_wmma_t<2, 8><<<blocks, 32, 0, stream>>>(A, B, bias, H); break;   //  64 -> 128
    case 3: gemm_wmma_t<5, 8><<<blocks, 32, 0, stream>>>(A, B, bias, H); break;   // 160 -> 128
    case 4: gemm_wmma_t<4, 8><<<blocks, 32, 0, stream>>>(A, B, bias, H); break;   // 128 -> 128
    case 5: gemm_wmma_t<4, 16><<<blocks, 32, 0, stream>>>(A, B, bias, H); break;  // 128 -> 256
  }
}

// ---------------------------------------------------------------------------
// host-side launch
// ---------------------------------------------------------------------------
extern "C" void kernel_launch(void* const* d_in, const int* in_sizes, int n_in,
                              void* d_out, int out_size, void* d_ws, size_t ws_size,
                              hipStream_t stream) {
  const float* x   = (const float*)d_in[0];   // [4,4096,64]
  const float* pos = (const float*)d_in[1];   // [4,4096,3]
  const float *W[6], *bia[6], *gam[6], *bet[6];
  for (int l = 0; l < 6; ++l) {
    W[l]   = (const float*)d_in[2 + l * 4 + 0];
    bia[l] = (const float*)d_in[2 + l * 4 + 1];
    gam[l] = (const float*)d_in[2 + l * 4 + 2];
    bet[l] = (const float*)d_in[2 + l * 4 + 3];
  }

  const int Bsz = 4, N0 = 4096, M1 = 2048, M2 = 512;
  const long long E1 = (long long)Bsz * M1 * KNB;   // 524288
  const long long E2 = (long long)Bsz * M2 * KNB;   // 131072
  const int Cin[6]  = {67, 64, 64, 131, 128, 128};
  const int Cpad[6] = {96, 64, 64, 160, 128, 128};
  const int Cout[6] = {64, 64, 128, 128, 128, 256};

  // workspace carve-up
  char*  base = (char*)d_ws;
  size_t off  = 0;
  auto alloc = [&](size_t bytes) -> void* {
    void* p = base + off;
    off = (off + bytes + 255) & ~(size_t)255;
    return p;
  };
  float*    centers1 = (float*)alloc((size_t)Bsz * M1 * 3 * 4);
  float*    centers2 = (float*)alloc((size_t)Bsz * M2 * 3 * 4);
  int*      nidx1    = (int*)  alloc((size_t)E1 * 4);
  int*      mask1    = (int*)  alloc((size_t)E1 * 4);
  int*      nidx2    = (int*)  alloc((size_t)E2 * 4);
  int*      mask2    = (int*)  alloc((size_t)E2 * 4);
  float*    cnts     = (float*)alloc(2 * 4);           // cnt1, cnt2
  float*    stats    = (float*)alloc(512 * 4);         // sum[256] + sumsq[256]
  _Float16* Wswz[6];
  for (int l = 0; l < 6; ++l) Wswz[l] = (_Float16*)alloc((size_t)Cout[l] * Cpad[l] * 2);
  float*    x1   = (float*)   alloc((size_t)Bsz * M1 * 128 * 4);
  _Float16* BUFA = (_Float16*)alloc((size_t)E1 * 96 * 2);    // max A: E1 x 96
  _Float16* BUFH = (_Float16*)alloc((size_t)E1 * 128 * 2);   // max H: E1 x 128

  // counters + weights
  zero_f32<<<1, 64, 0, stream>>>(cnts, 2);
  for (int l = 0; l < 6; ++l) {
    int tot = Cout[l] * Cpad[l];
    convert_w<<<(tot + 255) / 256, 256, 0, stream>>>(W[l], Cin[l], Cout[l], Cpad[l], Wswz[l]);
  }

  // ---------------- SA module 1 ----------------
  fps_kernel<<<Bsz, 1024, 0, stream>>>(pos, N0, M1, centers1);
  group_kernel<<<Bsz * M1, 64, 0, stream>>>(pos, centers1, N0, M1, 0.2f * 0.2f,
                                            nidx1, mask1, cnts + 0);
  build_edges<<<8192, 256, 0, stream>>>(x, 64, pos, centers1, nidx1, mask1,
                                        M1, N0, Cpad[0], E1, BUFA);
  for (int l = 0; l < 3; ++l) {
    launch_gemm(l, E1, (const v16h*)BUFA, (const v16h*)Wswz[l], bia[l], BUFH, stream);
    zero_f32<<<2, 256, 0, stream>>>(stats, 512);
    bn_stats<<<4096, 256, 0, stream>>>(BUFH, mask1, E1, Cout[l], stats, stats + 256);
    if (l < 2) {
      bn_apply<<<4096, 256, 0, stream>>>(BUFH, mask1, stats, stats + 256, cnts + 0,
                                         gam[l], bet[l], E1, Cout[l], BUFA);
    } else {
      bn_apply_max<<<Bsz * M1, 256, 0, stream>>>(BUFH, mask1, stats, stats + 256,
                                                 cnts + 0, gam[l], bet[l], Cout[l], x1);
    }
  }

  // ---------------- SA module 2 ----------------
  fps_kernel<<<Bsz, 1024, 0, stream>>>(centers1, M1, M2, centers2);
  group_kernel<<<Bsz * M2, 64, 0, stream>>>(centers1, centers2, M1, M2, 0.4f * 0.4f,
                                            nidx2, mask2, cnts + 1);
  build_edges<<<8192, 256, 0, stream>>>(x1, 128, centers1, centers2, nidx2, mask2,
                                        M2, M1, Cpad[3], E2, BUFA);
  float* x2_out = (float*)d_out;                        // [4,512,256]
  for (int l = 3; l < 6; ++l) {
    launch_gemm(l, E2, (const v16h*)BUFA, (const v16h*)Wswz[l], bia[l], BUFH, stream);
    zero_f32<<<2, 256, 0, stream>>>(stats, 512);
    bn_stats<<<4096, 256, 0, stream>>>(BUFH, mask2, E2, Cout[l], stats, stats + 256);
    if (l < 5) {
      bn_apply<<<4096, 256, 0, stream>>>(BUFH, mask2, stats, stats + 256, cnts + 1,
                                         gam[l], bet[l], E2, Cout[l], BUFA);
    } else {
      bn_apply_max<<<Bsz * M2, 256, 0, stream>>>(BUFH, mask2, stats, stats + 256,
                                                 cnts + 1, gam[l], bet[l], Cout[l], x2_out);
    }
  }

  // pos2 = centers2 appended after x2 in d_out
  int npos2 = Bsz * M2 * 3;
  copy_f32<<<(npos2 + 255) / 256, 256, 0, stream>>>(centers2,
                                                    (float*)d_out + (size_t)Bsz * M2 * 256,
                                                    npos2);
}
